// CausalSelfAttention_2482491097636
// MI455X (gfx1250) — matrix operand order repaired and checked by reading, not compiled, not run on hardware
//
#include <hip/hip_runtime.h>

// ---------------------------------------------------------------------------
// Types for CDNA5 WMMA (gfx1250, wave32)
// ---------------------------------------------------------------------------
typedef __bf16 bf16;
typedef __attribute__((ext_vector_type(8)))  __bf16 v8bf;
typedef __attribute__((ext_vector_type(16))) __bf16 v16bf;
typedef __attribute__((ext_vector_type(8)))  float  v8f;

#define N_HEAD  12
#define C_EMBD  768
#define T_SEQ   2048
#define B_BATCH 2
#define HD      64
#define NEGINF  (-1.0e30f)

__device__ __forceinline__ v8f wmma_bf16(v16bf a, v16bf b, v8f c) {
    // D = A(16x32) * B(32x16) + C(16x16), fp32 accumulate
    return __builtin_amdgcn_wmma_f32_16x16x32_bf16(
        /*neg_a=*/false, a, /*neg_b=*/false, b,
        /*c_mod=*/(short)0, c, /*reuse_a=*/false, /*reuse_b=*/false);
}

// A fragment: row-major A[M x K], lda = K. 16-bit A 16x32 layout (ISA 7.12.2):
//   lanes 0-15  : row = lane,    halves K=0..7  and K=16..23
//   lanes 16-31 : row = lane-16, halves K=8..15 and K=24..31
__device__ __forceinline__ v16bf load_a_frag(const bf16* A, int lda, int m0,
                                             int k0, int lane) {
    int row = lane & 15;
    int kh  = (lane >> 4) << 3;            // 0 or 8
    const bf16* p = A + (size_t)(m0 + row) * lda + (k0 + kh);
    v8bf lo = *(const v8bf*)p;             // K = kh .. kh+7
    v8bf hi = *(const v8bf*)(p + 16);      // K = kh+16 .. kh+23
    return __builtin_shufflevector(lo, hi, 0,1,2,3,4,5,6,7,8,9,10,11,12,13,14,15);
}

// B fragment: B stored K-contiguous per column, i.e. Bt[N][K], ldb = K.
//   lanes 0-15  : col = lane,    K=0..7  and K=16..23
//   lanes 16-31 : col = lane-16, K=8..15 and K=24..31
__device__ __forceinline__ v16bf load_b_frag(const bf16* Bt, int ldb, int n0,
                                             int k0, int lane) {
    int col = lane & 15;
    int kh  = (lane >> 4) << 3;            // 0 or 8
    const bf16* p = Bt + (size_t)(n0 + col) * ldb + (k0 + kh);
    v8bf lo = *(const v8bf*)p;
    v8bf hi = *(const v8bf*)(p + 16);
    return __builtin_shufflevector(lo, hi, 0,1,2,3,4,5,6,7,8,9,10,11,12,13,14,15);
}

// ---------------------------------------------------------------------------
// Elementwise prep kernels
// ---------------------------------------------------------------------------
__global__ void prep_xn_kernel(const float* __restrict__ x,
                               const float* __restrict__ sc,
                               const float* __restrict__ bi,
                               bf16* __restrict__ xn, int total, int C) {
    int idx = blockIdx.x * blockDim.x + threadIdx.x;
    if (idx >= total) return;
    int c = idx % C;
    xn[idx] = (bf16)(sc[c] * x[idx] + bi[c]);
}

// Wt[n*K + k] = bf16( W[k*N + n] )  (K-contiguous per output column)
__global__ void transpose_w_kernel(const float* __restrict__ W,
                                   bf16* __restrict__ Wt, int K, int N) {
    int idx = blockIdx.x * blockDim.x + threadIdx.x;
    if (idx >= K * N) return;
    int n = idx / K;
    int k = idx - n * K;
    Wt[idx] = (bf16)W[(size_t)k * N + n];
}

// ---------------------------------------------------------------------------
// bf16 WMMA GEMM: out[M,N] = A[M,K] @ Bt[N,K]^T + bias, optional *scale
// block = 256 (8 waves); wave tile = 16x64; WG tile = 128x64
// ---------------------------------------------------------------------------
__global__ void gemm_bf16_kernel(const bf16* __restrict__ A,
                                 const bf16* __restrict__ Bt,
                                 const float* __restrict__ bias,
                                 float* __restrict__ outf,
                                 bf16* __restrict__ outb,
                                 const float* __restrict__ scp,
                                 int M, int N, int K) {
    int lane = threadIdx.x & 31;
    int wave = threadIdx.x >> 5;
    int m0 = blockIdx.x * 128 + wave * 16;
    int n0 = blockIdx.y * 64;

    v8f acc0 = {}, acc1 = {}, acc2 = {}, acc3 = {};
    for (int k0 = 0; k0 < K; k0 += 32) {
        __builtin_prefetch(A + (size_t)(m0 + (lane & 15)) * K + k0 + 128, 0, 1);
        v16bf a  = load_a_frag(A, K, m0, k0, lane);
        v16bf b0 = load_b_frag(Bt, K, n0 +  0, k0, lane);
        v16bf b1 = load_b_frag(Bt, K, n0 + 16, k0, lane);
        v16bf b2 = load_b_frag(Bt, K, n0 + 32, k0, lane);
        v16bf b3 = load_b_frag(Bt, K, n0 + 48, k0, lane);
        acc0 = wmma_bf16(a, b0, acc0);
        acc1 = wmma_bf16(a, b1, acc1);
        acc2 = wmma_bf16(a, b2, acc2);
        acc3 = wmma_bf16(a, b3, acc3);
    }

    float sc = scp ? scp[0] : 1.0f;
    int colid  = lane & 15;
    int rowoff = (lane >> 4) * 8;
    v8f accs[4] = {acc0, acc1, acc2, acc3};
#pragma unroll
    for (int t = 0; t < 4; t++) {
        int n = n0 + t * 16 + colid;
        float bv = bias ? bias[n] : 0.0f;
#pragma unroll
        for (int r = 0; r < 8; r++) {
            int m = m0 + rowoff + r;
            float v = (accs[t][r] + bv) * sc;
            size_t oidx = (size_t)m * N + n;
            if (outb) outb[oidx] = (bf16)v;
            else      outf[oidx] = v;
        }
    }
}

// ---------------------------------------------------------------------------
// RoPE + split qkv[BT, 3C] -> q,k [BH,T,64] (rotated) and vt [BH,64,T]
// one thread per (b,t,h,pair), pair i handles dims 2i,2i+1
// ---------------------------------------------------------------------------
__global__ void rope_split_kernel(const bf16* __restrict__ qkv,
                                  bf16* __restrict__ q,
                                  bf16* __restrict__ k,
                                  bf16* __restrict__ vt) {
    int idx = blockIdx.x * blockDim.x + threadIdx.x;   // B*T*H*32 threads
    int i  = idx & 31;
    int h  = (idx >> 5) % N_HEAD;
    int bt = idx / (32 * N_HEAD);
    int t  = bt & (T_SEQ - 1);
    int b  = bt >> 11;

    const bf16* row = qkv + (size_t)bt * (3 * C_EMBD);
    int d = h * HD + 2 * i;
    float q0 = (float)row[d];
    float q1 = (float)row[d + 1];
    float k0 = (float)row[C_EMBD + d];
    float k1 = (float)row[C_EMBD + d + 1];
    float v0 = (float)row[2 * C_EMBD + d];
    float v1 = (float)row[2 * C_EMBD + d + 1];

    // freq_i = theta^(-2i/64); log(10000) = 9.210340371976184
    float freq = __expf(-(float)i * (9.210340371976184f / 32.0f));
    float ang = (float)t * freq;
    float s, c;
    __sincosf(ang, &s, &c);

    size_t bh = (size_t)(b * N_HEAD + h);
    size_t qo = (bh * T_SEQ + t) * HD + 2 * i;
    q[qo]     = (bf16)(q0 * c - q1 * s);
    q[qo + 1] = (bf16)(q0 * s + q1 * c);
    k[qo]     = (bf16)(k0 * c - k1 * s);
    k[qo + 1] = (bf16)(k0 * s + k1 * c);
    vt[(bh * HD + 2 * i) * T_SEQ + t]     = (bf16)v0;
    vt[(bh * HD + 2 * i + 1) * T_SEQ + t] = (bf16)v1;
}

// ---------------------------------------------------------------------------
// Flash attention with attention sink, causal, hd=64.
// grid = (T/64, B*H), block = 128 (4 waves). Wave owns 16 query rows.
// Per 32-key tile: 4 WMMA (Q*K^T) + 4 WMMA (P*V). P relaid C->A via LDS.
// ---------------------------------------------------------------------------
__global__ void flash_attn_kernel(const bf16* __restrict__ q,
                                  const bf16* __restrict__ k,
                                  const bf16* __restrict__ vt,
                                  const float* __restrict__ sinks,
                                  bf16* __restrict__ y) {
    __shared__ __align__(16) bf16 pshm[4][16 * 32];

    int lane = threadIdx.x & 31;
    int wave = threadIdx.x >> 5;               // 0..3
    int bh   = blockIdx.y;                     // 0..23
    int h    = bh % N_HEAD;
    int b    = bh / N_HEAD;
    int q0   = blockIdx.x * 64 + wave * 16;    // first query row of this wave

    const bf16* qh = q  + (size_t)bh * T_SEQ * HD;
    const bf16* kh = k  + (size_t)bh * T_SEQ * HD;
    const bf16* vh = vt + (size_t)bh * HD * T_SEQ;

    // Q fragments for hd = 64 (two K=32 chunks), reused across all key tiles
    v16bf qa0 = load_a_frag(qh, HD, q0, 0, lane);
    v16bf qa1 = load_a_frag(qh, HD, q0, 32, lane);

    int colid  = lane & 15;
    int rowoff = (lane >> 4) * 8;
    const float scale = 0.125f;                // 1/sqrt(64)

    float m_r[8], l_r[8];
#pragma unroll
    for (int r = 0; r < 8; r++) { m_r[r] = NEGINF; l_r[r] = 0.0f; }
    v8f o0 = {}, o1 = {}, o2 = {}, o3 = {};

    int kend = q0 + 16;                        // keys in [0, kend) may be valid
    for (int kt = 0; kt < kend; kt += 32) {
        // ---- S = Q * K^T for 16x32 key tile ----
        v16bf kb00 = load_b_frag(kh, HD, kt,      0, lane);
        v16bf kb01 = load_b_frag(kh, HD, kt,     32, lane);
        v16bf kb10 = load_b_frag(kh, HD, kt + 16, 0, lane);
        v16bf kb11 = load_b_frag(kh, HD, kt + 16, 32, lane);
        v8f s0 = {}, s1 = {};
        s0 = wmma_bf16(qa0, kb00, s0);
        s0 = wmma_bf16(qa1, kb01, s0);
        s1 = wmma_bf16(qa0, kb10, s1);
        s1 = wmma_bf16(qa1, kb11, s1);

        // ---- online softmax (per C-layout row slot) ----
#pragma unroll
        for (int r = 0; r < 8; r++) {
            int grow = q0 + rowoff + r;        // global query row
            float a = s0[r] * scale;
            float bb = s1[r] * scale;
            if (kt + colid      > grow) a  = NEGINF;
            if (kt + 16 + colid > grow) bb = NEGINF;

            float mx = fmaxf(a, bb);
            mx = fmaxf(mx, __shfl_xor(mx, 1, 32));
            mx = fmaxf(mx, __shfl_xor(mx, 2, 32));
            mx = fmaxf(mx, __shfl_xor(mx, 4, 32));
            mx = fmaxf(mx, __shfl_xor(mx, 8, 32));

            float mn = fmaxf(m_r[r], mx);
            float f  = __expf(m_r[r] - mn);
            float p0 = __expf(a  - mn);
            float p1 = __expf(bb - mn);

            float srow = p0 + p1;
            srow += __shfl_xor(srow, 1, 32);
            srow += __shfl_xor(srow, 2, 32);
            srow += __shfl_xor(srow, 4, 32);
            srow += __shfl_xor(srow, 8, 32);

            l_r[r] = l_r[r] * f + srow;
            m_r[r] = mn;
            o0[r] *= f; o1[r] *= f; o2[r] *= f; o3[r] *= f;

            // stash P (bf16) to this wave's LDS tile, row-major 16x32
            int prow = rowoff + r;
            pshm[wave][prow * 32 + colid]      = (bf16)p0;
            pshm[wave][prow * 32 + 16 + colid] = (bf16)p1;
        }

        // ---- read P back in A-fragment layout (16x32) ----
        v16bf pa;
        {
            int prow = lane & 15;
            int kh2  = (lane >> 4) << 3;
            const bf16* pp = &pshm[wave][prow * 32 + kh2];
            v8bf lo = *(const v8bf*)pp;
            v8bf hi = *(const v8bf*)(pp + 16);
            pa = __builtin_shufflevector(lo, hi,
                 0,1,2,3,4,5,6,7,8,9,10,11,12,13,14,15);
        }

        // ---- O += P * V (V transposed: contiguous in key dim) ----
        v16bf vb0 = load_b_frag(vh, T_SEQ,  0, kt, lane);
        v16bf vb1 = load_b_frag(vh, T_SEQ, 16, kt, lane);
        v16bf vb2 = load_b_frag(vh, T_SEQ, 32, kt, lane);
        v16bf vb3 = load_b_frag(vh, T_SEQ, 48, kt, lane);
        o0 = wmma_bf16(pa, vb0, o0);
        o1 = wmma_bf16(pa, vb1, o1);
        o2 = wmma_bf16(pa, vb2, o2);
        o3 = wmma_bf16(pa, vb3, o3);
    }

    // ---- sink contributes exp(sink - m) to the denominator only ----
    float sink = sinks[h];
#pragma unroll
    for (int r = 0; r < 8; r++) {
        float l = l_r[r] + __expf(sink - m_r[r]);
        float inv = 1.0f / l;
        int grow = q0 + rowoff + r;
        size_t base = ((size_t)(b * T_SEQ + grow)) * C_EMBD + h * HD;
        y[base +  0 + colid] = (bf16)(o0[r] * inv);
        y[base + 16 + colid] = (bf16)(o1[r] * inv);
        y[base + 32 + colid] = (bf16)(o2[r] * inv);
        y[base + 48 + colid] = (bf16)(o3[r] * inv);
    }
}

// ---------------------------------------------------------------------------
// Host launcher
// ---------------------------------------------------------------------------
extern "C" void kernel_launch(void* const* d_in, const int* in_sizes, int n_in,
                              void* d_out, int out_size, void* d_ws, size_t ws_size,
                              hipStream_t stream) {
    const float* x     = (const float*)d_in[0];
    const float* nsc   = (const float*)d_in[1];
    const float* nbi   = (const float*)d_in[2];
    const float* Wat   = (const float*)d_in[3];
    const float* bat   = (const float*)d_in[4];
    const float* Wpr   = (const float*)d_in[5];
    const float* bpr   = (const float*)d_in[6];
    const float* sinks = (const float*)d_in[7];
    const float* rez   = (const float*)d_in[8];
    float* out = (float*)d_out;

    const size_t BT = (size_t)B_BATCH * T_SEQ;           // 4096
    const size_t BHThd = (size_t)B_BATCH * N_HEAD * T_SEQ * HD;

    char* ws = (char*)d_ws;
    bf16* xn    = (bf16*)ws; ws += BT * C_EMBD * 2;                  // 6.3 MB
    bf16* wat_t = (bf16*)ws; ws += (size_t)3 * C_EMBD * C_EMBD * 2;  // 3.5 MB
    bf16* wpr_t = (bf16*)ws; ws += (size_t)C_EMBD * C_EMBD * 2;      // 1.2 MB
    bf16* qkv   = (bf16*)ws; ws += BT * 3 * C_EMBD * 2;              // 18.9 MB
    bf16* qb    = (bf16*)ws; ws += BHThd * 2;                        // 6.3 MB
    bf16* kb    = (bf16*)ws; ws += BHThd * 2;                        // 6.3 MB
    bf16* vtb   = (bf16*)ws; ws += BHThd * 2;                        // 6.3 MB
    bf16* yb    = (bf16*)ws; ws += BT * C_EMBD * 2;                  // 6.3 MB

    // 1) affine norm -> bf16
    {
        int total = (int)(BT * C_EMBD);
        prep_xn_kernel<<<(total + 255) / 256, 256, 0, stream>>>(
            x, nsc, nbi, xn, total, C_EMBD);
    }
    // 2) weight transposes -> bf16 K-contiguous
    {
        int t1 = C_EMBD * 3 * C_EMBD;
        transpose_w_kernel<<<(t1 + 255) / 256, 256, 0, stream>>>(
            Wat, wat_t, C_EMBD, 3 * C_EMBD);
        int t2 = C_EMBD * C_EMBD;
        transpose_w_kernel<<<(t2 + 255) / 256, 256, 0, stream>>>(
            Wpr, wpr_t, C_EMBD, C_EMBD);
    }
    // 3) QKV GEMM: [4096,2304] = xn @ W_attn + b_attn (bf16 out)
    gemm_bf16_kernel<<<dim3(32, 36), 256, 0, stream>>>(
        xn, wat_t, bat, nullptr, qkv, nullptr,
        (int)BT, 3 * C_EMBD, C_EMBD);
    // 4) RoPE + split/transposes
    {
        int total = B_BATCH * T_SEQ * N_HEAD * 32;
        rope_split_kernel<<<total / 256, 256, 0, stream>>>(qkv, qb, kb, vtb);
    }
    // 5) flash attention with sink
    flash_attn_kernel<<<dim3(T_SEQ / 64, B_BATCH * N_HEAD), 128, 0, stream>>>(
        qb, kb, vtb, sinks, yb);
    // 6) projection: out = (y @ W_proj + b_proj) * rezero  (fp32 out)
    gemm_bf16_kernel<<<dim3(32, 12), 256, 0, stream>>>(
        yb, wpr_t, bpr, out, nullptr, rez,
        (int)BT, C_EMBD, C_EMBD);

    (void)in_sizes; (void)n_in; (void)out_size; (void)ws_size;
}
